// GreedySampler_14791867367697
// MI455X (gfx1250) — matrix operand order repaired and checked by reading, not compiled, not run on hardware
//
#include <hip/hip_runtime.h>
#include <hip/hip_bf16.h>
#include <stdint.h>

#define NSEQ   32
#define DMODEL 4096
#define VOCAB  50257
#define TPW    2                 // 16-row vocab tiles per wave
#define ROWS_PER_WAVE (16 * TPW) // 32

typedef __attribute__((ext_vector_type(16))) __bf16 v16bf;
typedef __attribute__((ext_vector_type(8)))  float  v8f;

union BfPack { v16bf v; unsigned int u[8]; };

__device__ __forceinline__ unsigned int f2bf_rne(float f) {
  unsigned int u = __float_as_uint(f);
  u += 0x7FFFu + ((u >> 16) & 1u);   // round-to-nearest-even to bf16
  return u >> 16;
}
// pack two f32 -> two bf16 (round-to-nearest) in one v_perm_b32:
// result = (bf16(hi) << 16) | bf16(lo)
__device__ __forceinline__ unsigned int packbf(float lo, float hi) {
  unsigned int a = __float_as_uint(lo) + 0x8000u;   // round bias
  unsigned int b = __float_as_uint(hi) + 0x8000u;
  // concat bytes: {b[7:4], a[3:0]}; select a.byte2,a.byte3,b.byte2,b.byte3
  return __builtin_amdgcn_perm(b, a, 0x07060302u);
}
// monotone mapping: float -> uint32 preserving order
__device__ __forceinline__ unsigned int fkey(float f) {
  unsigned int u = __float_as_uint(f);
  return (u & 0x80000000u) ? ~u : (u | 0x80000000u);
}

// ---------------- kernel 1: cumsum indices + init accumulators ---------------
__global__ void prep_kernel(const int* __restrict__ lens,
                            unsigned long long* __restrict__ best,
                            int* __restrict__ idx) {
  int t = threadIdx.x;
  if (t < NSEQ) {
    int s = 0;
    for (int j = 0; j <= t; ++j) s += lens[j];
    idx[t]  = s - 1;
    best[t] = 0ull;
  }
}

// ---------------- kernel 2: gather last-token rows, convert to bf16 ----------
__global__ void gather_kernel(const float* __restrict__ hidden,
                              const int* __restrict__ idx,
                              unsigned short* __restrict__ hbf) {
  int k   = blockIdx.x * blockDim.x + threadIdx.x;  // 0..4095
  int seq = blockIdx.y;                             // 0..31
  int row = idx[seq];
  hbf[(size_t)seq * DMODEL + k] =
      (unsigned short)f2bf_rne(hidden[(size_t)row * DMODEL + k]);
}

// ---------------- kernel 3: WMMA GEMM + fused per-seq argmax -----------------
__global__ __launch_bounds__(256)
void gemm_argmax_kernel(const float* __restrict__ W,
                        const unsigned short* __restrict__ hbf,
                        unsigned long long* __restrict__ best) {
  const int lane = threadIdx.x & 31;
  const int wave = blockIdx.x * (blockDim.x >> 5) + (threadIdx.x >> 5);
  const int half = lane >> 4;   // 0: lanes 0-15, 1: lanes 16-31
  const int lrow = lane & 15;
  const int base_row = wave * ROWS_PER_WAVE;
  if (base_row >= VOCAB) return;

  // weight row pointers for the tiles (clamped for OOB rows)
  const float* wrow[TPW];
#pragma unroll
  for (int t = 0; t < TPW; ++t) {
    int m  = base_row + t * 16 + lrow;
    int mc = m < VOCAB ? m : (VOCAB - 1);
    wrow[t] = W + (size_t)mc * DMODEL;
  }

  v8f acc[TPW][2];
#pragma unroll
  for (int t = 0; t < TPW; ++t)
#pragma unroll
    for (int g = 0; g < 2; ++g)
#pragma unroll
      for (int r = 0; r < 8; ++r) acc[t][g][r] = 0.0f;

  // B operand source rows: seq = lrow (group0), lrow+16 (group1);
  // lane covers K = ks + half*16 .. +15 (contiguous k order, same as A)
  const unsigned short* hb0 = hbf + (size_t)(lrow)      * DMODEL + half * 16;
  const unsigned short* hb1 = hbf + (size_t)(lrow + 16) * DMODEL + half * 16;

#pragma unroll 2
  for (int ks = 0; ks < DMODEL; ks += 32) {
    // ---- issue all loads for this K-step first ----
    const uint4* p0 = (const uint4*)(hb0 + ks);
    const uint4* p1 = (const uint4*)(hb1 + ks);
    uint4 q0 = p0[0], q1 = p0[1];
    uint4 r0 = p1[0], r1 = p1[1];

    float4 a[TPW][4];
#pragma unroll
    for (int t = 0; t < TPW; ++t) {
      const float* rp = wrow[t] + ks + half * 16;       // K run [h*16, h*16+8)
      const float* rq = wrow[t] + ks + half * 16 + 8;   // K run [h*16+8, h*16+16)
      a[t][0] = ((const float4*)rp)[0];
      a[t][1] = ((const float4*)rp)[1];
      a[t][2] = ((const float4*)rq)[0];
      a[t][3] = ((const float4*)rq)[1];
      __builtin_prefetch(wrow[t] + ks + 64, 0, 0);      // global_prefetch_b8
    }

    // ---- build operands (v_perm packing) + WMMA ----
    BfPack B0, B1;
    B0.u[0] = q0.x; B0.u[1] = q0.y; B0.u[2] = q0.z; B0.u[3] = q0.w;
    B0.u[4] = q1.x; B0.u[5] = q1.y; B0.u[6] = q1.z; B0.u[7] = q1.w;
    B1.u[0] = r0.x; B1.u[1] = r0.y; B1.u[2] = r0.z; B1.u[3] = r0.w;
    B1.u[4] = r1.x; B1.u[5] = r1.y; B1.u[6] = r1.z; B1.u[7] = r1.w;

#pragma unroll
    for (int t = 0; t < TPW; ++t) {
      BfPack A;
      A.u[0] = packbf(a[t][0].x, a[t][0].y);
      A.u[1] = packbf(a[t][0].z, a[t][0].w);
      A.u[2] = packbf(a[t][1].x, a[t][1].y);
      A.u[3] = packbf(a[t][1].z, a[t][1].w);
      A.u[4] = packbf(a[t][2].x, a[t][2].y);
      A.u[5] = packbf(a[t][2].z, a[t][2].w);
      A.u[6] = packbf(a[t][3].x, a[t][3].y);
      A.u[7] = packbf(a[t][3].z, a[t][3].w);
      acc[t][0] = __builtin_amdgcn_wmma_f32_16x16x32_bf16(
          false, A.v, false, B0.v, (short)0, acc[t][0], false, false);
      acc[t][1] = __builtin_amdgcn_wmma_f32_16x16x32_bf16(
          false, A.v, false, B1.v, (short)0, acc[t][1], false, false);
    }
  }

  // D layout: elem r on lanes 0-15 -> M=r, lanes 16-31 -> M=8+r; N = lrow
  unsigned long long loc0 = 0ull, loc1 = 0ull;
#pragma unroll
  for (int t = 0; t < TPW; ++t) {
    int vbase = base_row + t * 16 + half * 8;
#pragma unroll
    for (int r = 0; r < 8; ++r) {
      int v = vbase + r;
      if (v < VOCAB) {
        unsigned long long k0 =
            ((unsigned long long)fkey(acc[t][0][r]) << 32) | (unsigned int)(~v);
        unsigned long long k1 =
            ((unsigned long long)fkey(acc[t][1][r]) << 32) | (unsigned int)(~v);
        if (k0 > loc0) loc0 = k0;
        if (k1 > loc1) loc1 = k1;
      }
    }
  }
  // combine the two half-wave groups (same seq, different vocab rows)
  unsigned long long o0 = __shfl_xor(loc0, 16, 32);
  unsigned long long o1 = __shfl_xor(loc1, 16, 32);
  if (o0 > loc0) loc0 = o0;
  if (o1 > loc1) loc1 = o1;
  if (half == 0) {
    atomicMax(&best[lrow],      loc0);  // seqs 0..15
    atomicMax(&best[lrow + 16], loc1);  // seqs 16..31
  }
}

// ---------------- kernel 4: unpack winners to int32 ids ----------------------
__global__ void final_kernel(const unsigned long long* __restrict__ best,
                             int* __restrict__ out) {
  int t = threadIdx.x;
  if (t < NSEQ)
    out[t] = (int)(~(unsigned int)(best[t] & 0xFFFFFFFFull));
}

extern "C" void kernel_launch(void* const* d_in, const int* in_sizes, int n_in,
                              void* d_out, int out_size, void* d_ws, size_t ws_size,
                              hipStream_t stream) {
  const float* hidden = (const float*)d_in[0];   // [16384, 4096] f32
  const float* W      = (const float*)d_in[1];   // [50257, 4096] f32
  const int*   lens   = (const int*)d_in[2];     // [32] i32

  // workspace layout
  unsigned long long* best = (unsigned long long*)d_ws;                   // 256 B
  int*            idx  = (int*)((char*)d_ws + 256);                       // 128 B
  unsigned short* hbf  = (unsigned short*)((char*)d_ws + 1024);           // 256 KB

  prep_kernel<<<1, 32, 0, stream>>>(lens, best, idx);
  gather_kernel<<<dim3(DMODEL / 256, NSEQ), 256, 0, stream>>>(hidden, idx, hbf);

  int waves  = (VOCAB + ROWS_PER_WAVE - 1) / ROWS_PER_WAVE;  // 1571
  int blocks = (waves + 7) / 8;                              // 8 waves / block
  gemm_argmax_kernel<<<blocks, 256, 0, stream>>>(W, hbf, best);

  final_kernel<<<1, 32, 0, stream>>>(best, (int*)d_out);
}
